// BiGLSTM_57277683859536
// MI455X (gfx1250) — matrix operand order repaired and checked by reading, not compiled, not run on hardware
//
#include <hip/hip_runtime.h>
#include <hip/hip_bf16.h>
#include <math.h>

// ---------------------------------------------------------------------------
// BiGLSTM for MI455X (gfx1250): bf16 WMMA (fp32 accumulate) GEMMs + fp32 cell.
// T=32, N=1024, F=64, H=128, GNN_LAYERS=2. Only the final carry of each
// direction feeds the readout (reference only uses out[-1]).
// All GEMM strides / K depths are compile-time template params so global
// loads use immediate offsets (clause-able) instead of per-load v_add_u64.
// ---------------------------------------------------------------------------

#define TT_ 32
#define NN_ 1024
#define FF_ 64
#define HH_ 128
#define NH_ (NN_ * HH_)     // 131072
#define NZ_ (NN_ * 4 * HH_) // 524288

typedef __attribute__((ext_vector_type(16))) __bf16 v16bf;
typedef __attribute__((ext_vector_type(8)))  float  v8f;
typedef __attribute__((ext_vector_type(4)))  float  fvec4; // trivial, AS-friendly

// ---- addrspace(1) load/store helpers: force global_* (not flat_*) ----------
typedef const __attribute__((address_space(1))) float  GF;
typedef const __attribute__((address_space(1))) fvec4  GF4;
typedef __attribute__((address_space(1))) float        GFw;

__device__ __forceinline__ float ldg1(const float* p) {
    return *(GF*)(unsigned long long)p;
}
__device__ __forceinline__ fvec4 ldg4(const float* p) {
    return *(GF4*)(unsigned long long)p;
}
__device__ __forceinline__ void stg1(float* p, float v) {
    *(GFw*)(unsigned long long)p = v;
}

// ---------------------------------------------------------------------------
// One K-partition of the GEMM: acc += A[M x K] * B[K x Ncols] (bf16 WMMA).
// Block: 256 threads = 8 waves; tile 128(M) x 128(N); K-step 32.
// Wave w owns rows m0 = blockY*128 + 16w (16 rows x 128 cols -> 8 WMMA tiles).
//
// B staging: thread t handles column bn = t&127 of the 32x128 tile and
// k-half bkh = (t>>7)*16. It reads a 16-deep K column (each k iteration is a
// fully coalesced 512B read across the 128 threads of the half-block),
// converts to bf16, and writes ONE contiguous 32B LDS chunk that is exactly
// the v16bf B-fragment for (subtile j = bn>>4, lane = (bn&15) + bkh).
// B-fragment order used here (dense 32x16 KxN, contiguous k): lanes 0-15 hold K=0..15 of column
// N=lane; lanes 16-31 hold K=16..31; element e <-> K = (lane>=16)*16 + e.
// ---------------------------------------------------------------------------
template <int LDA, int LDB, int K>
__device__ __forceinline__ void gemm_part(
    const float* __restrict__ A, const float* __restrict__ B,
    const int n0, const int m0, const int tid,
    __bf16* __restrict__ Bsh, v8f* __restrict__ acc)
{
    const int lane = tid & 31;
    const int l15  = lane & 15;
    const int hi   = lane >> 4;

    const int bn    = tid & 127;        // B column within tile
    const int bkh   = (tid >> 7) << 4;  // 0 or 16: which K half
    const int bj    = bn >> 4;
    const int blane = (bn & 15) + bkh;
    __bf16* const bdst = &Bsh[((bj << 5) + blane) << 4];

    // Loop-invariant bases; per-iteration addresses fold into immediates.
    const float* const ap_base = A + (size_t)(m0 + l15) * LDA + hi * 16;
    const float* const bp_base = B + (size_t)bkh * LDB + (n0 + bn);

    for (int k0 = 0; k0 < K; k0 += 32) {
        // ---- A fragment loads: issue before the barrier so they are in
        //      flight while B is staged. A-fragment order used here (16x32,
        //      contiguous k): lanes 0-15 -> row M=L, K=0..15; lanes 16-31
        //      -> row M=L-16, K=16..31  => per-lane offset hi*16.
        const float* ap = ap_base + k0;
        const fvec4 a0 = ldg4(ap);
        const fvec4 a1 = ldg4(ap + 4);
        const fvec4 a2 = ldg4(ap + 8);
        const fvec4 a3 = ldg4(ap + 12);

        __syncthreads(); // previous iteration done reading Bsh

        // ---- stage B tile (32 x 128) -> LDS bf16, fragment order ----
        {
            const float* bp = bp_base + (size_t)k0 * LDB;
            v16bf frag;
#pragma unroll
            for (int i = 0; i < 16; i++)
                frag[i] = (__bf16)ldg1(bp + i * LDB); // immediate offsets
            *(v16bf*)bdst = frag; // one contiguous 32B chunk
        }
        __syncthreads();

        // ---- build A fragment (fp32 -> bf16) ----
        v16bf af;
#pragma unroll
        for (int i = 0; i < 4; i++) {
            af[i]      = (__bf16)a0[i];
            af[4 + i]  = (__bf16)a1[i];
            af[8 + i]  = (__bf16)a2[i];
            af[12 + i] = (__bf16)a3[i];
        }

        // ---- preload all 8 B fragments, then the WMMA chain ----
        v16bf bf[8];
#pragma unroll
        for (int j = 0; j < 8; j++)
            bf[j] = *(const v16bf*)&Bsh[((j << 5) + lane) << 4];
#pragma unroll
        for (int j = 0; j < 8; j++)
            acc[j] = __builtin_amdgcn_wmma_f32_16x16x32_bf16(
                /*neg_a=*/false, af, /*neg_b=*/false, bf[j],
                /*c_mod=*/(short)0, acc[j], /*reuse_a=*/false, /*reuse_b=*/false);
    }
}

template <int LDC, bool HASBIAS>
__device__ __forceinline__ void gemm_epilogue(
    const float* __restrict__ bias, float* __restrict__ C,
    const int n0, const int m0, const int tid, const v8f* __restrict__ acc)
{
    const int lane = tid & 31;
    const int l15  = lane & 15;
    const int hi   = lane >> 4;
    // C layout (8 VGPRs): elem r, lane L -> row m0 + r + hi*8, col n0+16j+l15.
#pragma unroll
    for (int j = 0; j < 8; j++) {
        const int col = n0 + (j << 4) + l15;
        const float bv = HASBIAS ? ldg1(bias + col) : 0.0f;
        float* cp = C + (size_t)(m0 + hi * 8) * LDC + col;
#pragma unroll
        for (int r = 0; r < 8; r++)
            stg1(cp + r * LDC, acc[j][r] + bv);
    }
}

// Single-part GEMM: C = A*B (+bias)
template <int LDA, int LDB, int K, int LDC, bool HASBIAS>
__global__ __launch_bounds__(256)
void gemm1_kernel(const float* __restrict__ A, const float* __restrict__ B,
                  const float* __restrict__ bias, float* __restrict__ C)
{
    __shared__ __attribute__((aligned(32))) __bf16 Bsh[8 * 32 * 16]; // 8KB
    const int tid = threadIdx.x;
    const int n0  = blockIdx.x * 128;
    const int m0  = blockIdx.y * 128 + (tid >> 5) * 16;

    v8f acc[8];
#pragma unroll
    for (int j = 0; j < 8; j++)
#pragma unroll
        for (int e = 0; e < 8; e++) acc[j][e] = 0.0f;

    gemm_part<LDA, LDB, K>(A, B, n0, m0, tid, Bsh, acc);
    gemm_epilogue<LDC, HASBIAS>(bias, C, n0, m0, tid, acc);
}

// Fused 3-part GEMM (the GLSTM cell): C = A0*B0 + A1*B1 + A2*B2 + bias
template <int LDA, int LDB, int K, int LDC>
__global__ __launch_bounds__(256)
void gemm3_kernel(const float* __restrict__ A0, const float* __restrict__ B0,
                  const float* __restrict__ A1, const float* __restrict__ B1,
                  const float* __restrict__ A2, const float* __restrict__ B2,
                  const float* __restrict__ bias, float* __restrict__ C)
{
    __shared__ __attribute__((aligned(32))) __bf16 Bsh[8 * 32 * 16]; // 8KB
    const int tid = threadIdx.x;
    const int n0  = blockIdx.x * 128;
    const int m0  = blockIdx.y * 128 + (tid >> 5) * 16;

    v8f acc[8];
#pragma unroll
    for (int j = 0; j < 8; j++)
#pragma unroll
        for (int e = 0; e < 8; e++) acc[j][e] = 0.0f;

    gemm_part<LDA, LDB, K>(A0, B0, n0, m0, tid, Bsh, acc);
    gemm_part<LDA, LDB, K>(A1, B1, n0, m0, tid, Bsh, acc);
    gemm_part<LDA, LDB, K>(A2, B2, n0, m0, tid, Bsh, acc);
    gemm_epilogue<LDC, true>(bias, C, n0, m0, tid, acc);
}

// ---------------------------------------------------------------------------
// Pack Win_w (H x F, row-major) -> WinT (F x H, row-major): WinT[k][n] = Win_w[n][k]
// ---------------------------------------------------------------------------
__global__ __launch_bounds__(256)
void packT_kernel(const float* __restrict__ src, float* __restrict__ dst)
{
    const int i = blockIdx.x * 256 + threadIdx.x; // 0 .. F*H-1
    const int k = i >> 7;       // 0..F-1
    const int n = i & 127;      // 0..H-1
    stg1(&dst[i], ldg1(&src[n * FF_ + k]));
}

// ---------------------------------------------------------------------------
// LSTM gate elementwise: z (N x 4H, order i|f|o|g), c_in (N x H) -> h_out,c_out
// ---------------------------------------------------------------------------
__global__ __launch_bounds__(256)
void gates_kernel(const float* __restrict__ z, const float* __restrict__ c_in,
                  float* __restrict__ h_out, float* __restrict__ c_out)
{
    const int idx  = blockIdx.x * 256 + threadIdx.x;  // 0..N*H-1
    const int node = idx >> 7;
    const int hh   = idx & 127;
    const float* zr = z + (size_t)node * (4 * HH_);
    const float gi = ldg1(zr + hh);
    const float gf = ldg1(zr + HH_ + hh);
    const float go = ldg1(zr + 2 * HH_ + hh);
    const float gg = ldg1(zr + 3 * HH_ + hh);
    const float si = 1.0f / (1.0f + __expf(-gi));
    const float sf = 1.0f / (1.0f + __expf(-gf));
    const float so = 1.0f / (1.0f + __expf(-go));
    const float c2 = sf * ldg1(c_in + idx) + si * tanhf(gg);
    const float h2 = so * tanhf(c2);
    stg1(c_out + idx, c2);
    stg1(h_out + idx, h2);
}

// Initialize carry: h_time = c_time = first frame of the direction's sequence.
__global__ __launch_bounds__(256)
void copy2_kernel(const float* __restrict__ src, float* __restrict__ d1,
                  float* __restrict__ d2)
{
    const int i = blockIdx.x * 256 + threadIdx.x;
    const float v = ldg1(src + i);
    stg1(d1 + i, v);
    stg1(d2 + i, v);
}

// ---------------------------------------------------------------------------
// Fused readout: y = (concat(hf,hb) @ fc0_w.T + fc0_b) @ wout_w.T + wout_b
// One block per node, 128 threads (one per fc0 channel), fp32 exact.
// ---------------------------------------------------------------------------
__global__ __launch_bounds__(128)
void readout_kernel(const float* __restrict__ hf, const float* __restrict__ hb,
                    const float* __restrict__ fc0_w, const float* __restrict__ fc0_b,
                    const float* __restrict__ wout_w, const float* __restrict__ wout_b,
                    float* __restrict__ y)
{
    const int node = blockIdx.x;
    const int tid  = threadIdx.x; // 0..127
    __shared__ float sh[2 * HH_];
    __shared__ float red[HH_];

    sh[tid]       = ldg1(hf + (size_t)node * HH_ + tid);
    sh[HH_ + tid] = ldg1(hb + (size_t)node * HH_ + tid);
    __syncthreads();

    const float* w = fc0_w + (size_t)tid * (2 * HH_);
    float acc = ldg1(fc0_b + tid);
#pragma unroll 8
    for (int k = 0; k < 2 * HH_; k++) acc += sh[k] * ldg1(w + k);

    red[tid] = acc * ldg1(wout_w + tid);
    __syncthreads();
    for (int s = 64; s > 0; s >>= 1) {
        if (tid < s) red[tid] += red[tid + s];
        __syncthreads();
    }
    if (tid == 0) stg1(y + node, red[0] + ldg1(wout_b));
}

// ---------------------------------------------------------------------------
// Host orchestration
// ---------------------------------------------------------------------------
extern "C" void kernel_launch(void* const* d_in, const int* in_sizes, int n_in,
                              void* d_out, int out_size, void* d_ws, size_t ws_size,
                              hipStream_t stream)
{
    (void)in_sizes; (void)n_in; (void)out_size; (void)ws_size;

    const float* x      = (const float*)d_in[0];   // (1,T,N,F)
    const float* adjs   = (const float*)d_in[1];   // (1,T,N,N)
    const float* Win_w  = (const float*)d_in[3];   // (H,F)
    const float* Win_b  = (const float*)d_in[4];   // (H)
    const float* Wx[2]  = {(const float*)d_in[5],  (const float*)d_in[9]};   // (H,4H)
    const float* Wh[2]  = {(const float*)d_in[6],  (const float*)d_in[10]};
    const float* Wn[2]  = {(const float*)d_in[7],  (const float*)d_in[11]};
    const float* bgate[2] = {(const float*)d_in[8], (const float*)d_in[12]}; // (4H)
    const float* fc0_w  = (const float*)d_in[13];  // (H,2H)
    const float* fc0_b  = (const float*)d_in[14];
    const float* wout_w = (const float*)d_in[15];  // (1,H)
    const float* wout_b = (const float*)d_in[16];
    float* y = (float*)d_out;                      // (N)

    // Workspace layout (fp32): xs | WinT | per-dir {h,c,htmp,ctmp,nmsg,z}
    float* ws = (float*)d_ws;
    float* xs = ws;
    size_t off = (size_t)TT_ * NH_;
    float* WinT = ws + off; off += (size_t)FF_ * HH_;
    float *ht[2], *ct[2], *htmp[2], *ctmp[2], *nm[2], *zz[2];
    for (int d = 0; d < 2; d++) {
        ht[d]   = ws + off; off += NH_;
        ct[d]   = ws + off; off += NH_;
        htmp[d] = ws + off; off += NH_;
        ctmp[d] = ws + off; off += NH_;
        nm[d]   = ws + off; off += NH_;
        zz[d]   = ws + off; off += NZ_;
    }

    const dim3 blk(256);

    // 0) Pack Win_w.T once (F x H)
    packT_kernel<<<(FF_ * HH_) / 256, blk, 0, stream>>>(Win_w, WinT);

    // 1) Input projection: xs = x @ Win_w.T + Win_b   (M=T*N, K=F, Ncols=H)
    gemm1_kernel<FF_, HH_, FF_, HH_, true>
        <<<dim3(1, (TT_ * NN_) / 128), blk, 0, stream>>>(x, WinT, Win_b, xs);

    // 2) Init carries: forward from xs[0], backward from xs[T-1]
    copy2_kernel<<<NH_ / 256, blk, 0, stream>>>(xs, ht[0], ct[0]);
    copy2_kernel<<<NH_ / 256, blk, 0, stream>>>(xs + (size_t)(TT_ - 1) * NH_, ht[1], ct[1]);

    // 3) Sequential scan (both directions interleaved per step)
    for (int t = 0; t < TT_; t++) {
        for (int d = 0; d < 2; d++) {
            const int tt = (d == 0) ? t : (TT_ - 1 - t);
            const float* xt  = xs + (size_t)tt * NH_;
            const float* adj = adjs + (size_t)tt * NN_ * NN_;

            // nmsg = adj @ h_time   (M=N, K=N, Ncols=H)
            gemm1_kernel<NN_, HH_, NN_, HH_, false>
                <<<dim3(1, NN_ / 128), blk, 0, stream>>>(adj, ht[d], nullptr, nm[d]);

            // layer 0: z = xt@Wx + h_time@Wh + nmsg@Wn + b  (M=N, Ncols=4H)
            gemm3_kernel<HH_, 4 * HH_, HH_, 4 * HH_>
                <<<dim3(4, NN_ / 128), blk, 0, stream>>>(
                    xt, Wx[d], ht[d], Wh[d], nm[d], Wn[d], bgate[d], zz[d]);
            gates_kernel<<<NH_ / 256, blk, 0, stream>>>(zz[d], ct[d], htmp[d], ctmp[d]);

            // layer 1 (nmsg reused: reference passes carry h_time to both layers)
            gemm3_kernel<HH_, 4 * HH_, HH_, 4 * HH_>
                <<<dim3(4, NN_ / 128), blk, 0, stream>>>(
                    xt, Wx[d], htmp[d], Wh[d], nm[d], Wn[d], bgate[d], zz[d]);
            gates_kernel<<<NH_ / 256, blk, 0, stream>>>(zz[d], ctmp[d], ht[d], ct[d]);
        }
    }

    // 4) Readout from final carries only (out[-1] is all the reference uses)
    readout_kernel<<<NN_, dim3(128), 0, stream>>>(
        ht[0], ht[1], fc0_w, fc0_b, wout_w, wout_b, y);
}
